// NGAMOE_33870112096348
// MI455X (gfx1250) — matrix operand order repaired and checked by reading, not compiled, not run on hardware
//
#include <hip/hip_runtime.h>

// Problem constants (from the reference)
#define Bsz   2048
#define Fdim  64
#define Wdim  256
#define Kdim  16
#define Odim  32
#define BT    128     // batch rows per workgroup
#define NTH   256     // 8 wave32 waves per workgroup
#define GSTRIDE 132   // padded row stride (floats) for transposed g buffer

typedef __attribute__((ext_vector_type(16))) __bf16 bf16x16;
typedef __attribute__((ext_vector_type(8)))  float  f32x8;

union ABfrag {
  bf16x16  v;
  uint4    q[2];
  unsigned u[8];
};

// f32 -> bf16 bits, round-to-nearest-even (pure integer, no __bf16 arithmetic)
__device__ __forceinline__ unsigned short f2bf(float x) {
  union { float f; unsigned u; } v; v.f = x;
  unsigned r = v.u + 0x7FFFu + ((v.u >> 16) & 1u);
  return (unsigned short)(r >> 16);
}
__device__ __forceinline__ unsigned pack2bf(float lo, float hi) {
  return (unsigned)f2bf(lo) | ((unsigned)f2bf(hi) << 16);
}

// Stage a [Wdim x ncols] f32 row-major weight matrix into LDS, pre-swizzled into
// WMMA B-fragment-major order with kt fastest (so the k-loop uses DS immediates):
//   fragment (ct, kt) at dword offset (ct*8 + kt)*256; lane (h*16+l16) owns
//   dwords [lane*8, lane*8+8) = K-pairs p = kt*16 + (j>>2)*8 + h*4 + (j&3) at column ct*16+l16.
// Compute reads each fragment as 2x ds_load_b128 per lane, conflict-free.
__device__ __forceinline__ void stage_weights(const float* __restrict__ Wg,
                                              unsigned* __restrict__ WS,
                                              int ncols, int tid) {
  const int total = (Wdim / 2) * ncols;      // output dwords
  for (int i = tid; i < total; i += NTH) {
    const int p = i / ncols;                 // K-pair index (rows 2p, 2p+1)
    const int c = i - p * ncols;             // output column
    const float a = Wg[(2 * p) * ncols + c];
    const float b = Wg[(2 * p + 1) * ncols + c];
    const int kt  = p >> 4;
    const int h   = (p >> 2) & 1;
    const int j   = (p & 3) | ((p >> 1) & 4);
    const int ct  = c >> 4;
    const int l16 = c & 15;
    WS[((((ct << 3) + kt) << 5) + (h << 4) + l16) * 8 + j] = pack2bf(a, b);
  }
}

// Warm L2 for the next weight matrix (global_prefetch_b8)
__device__ __forceinline__ void prefetch_weights(const float* p, int tid) {
  __builtin_prefetch(p + (tid << 4), 0, 1);
  __builtin_prefetch(p + 4096 + (tid << 4), 0, 1);
  __builtin_prefetch(p + 8192 + (tid << 4), 0, 1);
  __builtin_prefetch(p + 12288 + (tid << 4), 0, 1);
}

// h0[row][w] = relu(x[row] * Wi[w] + bi[w]); store bf16-packed into LDS
__device__ __forceinline__ void input_layer(const float* __restrict__ xcol,  // stride Fdim
                                            const float* __restrict__ Wi,
                                            const float* __restrict__ bi,
                                            unsigned short* __restrict__ H, int tid) {
  unsigned* Hd = (unsigned*)H;
  for (int i = tid; i < BT * (Wdim / 2); i += NTH) {
    const int row = i >> 7;       // / (Wdim/2)
    const int wp  = i & 127;
    const float xv = xcol[row * Fdim];
    const int w = wp * 2;
    const float a = fmaxf(fmaf(xv, Wi[w],     bi[w]),     0.f);
    const float b = fmaxf(fmaf(xv, Wi[w + 1], bi[w + 1]), 0.f);
    Hd[i] = pack2bf(a, b);
  }
}

// Hout = relu(Hin[BT x Wdim](bf16) @ Wslab(swizzled bf16) + bias)
// STORE_BF16: Hout is bf16 [BT x ncols]. Else: Hout is f32 transposed
// [ncols x GSTRIDE] (used only for the exp-path g buffer, ncols == Kdim).
// Each wave owns row-tile rt = wave (16 rows), loops over ncols/16 col tiles.
template <bool STORE_BF16>
__device__ __forceinline__ void gemm_layer(const unsigned short* __restrict__ Hin,
                                           void* __restrict__ Hout,
                                           const unsigned* __restrict__ WS,
                                           const float* __restrict__ bias,
                                           int ncols, int wave, int lane) {
  const int rt  = wave;
  const int l16 = lane & 15;
  const int kh  = (lane >> 4) * 8;   // A-frag: K sub-offset per lane-half

  // Hoist the wave's A fragments (16 rows x K=256) once: 2 x b128 per k-slab.
  ABfrag A[8];
  const unsigned short* arow = Hin + (rt * 16 + l16) * Wdim;
#pragma unroll
  for (int kt = 0; kt < 8; ++kt) {
    A[kt].q[0] = *(const uint4*)(arow + kt * 32 + kh);
    A[kt].q[1] = *(const uint4*)(arow + kt * 32 + 16 + kh);
  }

  const int nct = ncols >> 4;
  const unsigned* wlane = WS + (lane << 3);   // this lane's chunk inside a fragment
  for (int ct = 0; ct < nct; ++ct) {
    const int col = ct * 16 + l16;
    const float bv = bias[col];
    const unsigned* wct = wlane + (ct << 11);    // ct*8 fragments * 256 dwords
    f32x8 C = {0.f, 0.f, 0.f, 0.f, 0.f, 0.f, 0.f, 0.f};
#pragma unroll
    for (int kt = 0; kt < 8; ++kt) {
      ABfrag Bf;
      const uint4* wp = (const uint4*)(wct + (kt << 8));  // +kt*1KB: DS immediate
      Bf.q[0] = wp[0];
      Bf.q[1] = wp[1];
      C = __builtin_amdgcn_wmma_f32_16x16x32_bf16(false, A[kt].v, false, Bf.v,
                                                  (short)0, C, false, false);
    }
    // bias + relu + store (C layout: VGPR v -> row v + 8*(lane/16), col = lane%16)
    const int rbase = rt * 16 + ((lane >> 4) * 8);
#pragma unroll
    for (int v = 0; v < 8; ++v) {
      const float val = fmaxf(C[v] + bv, 0.f);
      const int r = rbase + v;
      if (STORE_BF16) ((unsigned short*)Hout)[r * ncols + col] = f2bf(val);
      else            ((float*)Hout)[col * GSTRIDE + r] = val;   // transposed g
    }
  }
}

// Full per-feature MLP: input layer -> 2 hidden (256x256) -> output (256 x ncols_out)
template <bool OUT_BF16>
__device__ __forceinline__ void run_path(const float* xcol,
                                         const float* Wi, const float* bi,
                                         const float* Wh, const float* bh,
                                         const float* Wo, const float* bo,
                                         int ncols_out,
                                         unsigned short* HA, unsigned short* HB,
                                         unsigned* WS, void* OutBuf,
                                         int f, int tid, int wave, int lane) {
  input_layer(xcol, Wi + f * Wdim, bi + f * Wdim, HA, tid);
  prefetch_weights(Wh + ((size_t)f) * (Wdim * Wdim), tid);
  __syncthreads();

  const unsigned short* cur = HA;
  unsigned short* nxt = HB;
#pragma unroll
  for (int l = 0; l < 2; ++l) {
    stage_weights(Wh + ((size_t)(l * Fdim + f)) * (Wdim * Wdim), WS, Wdim, tid);
    if (l == 0) prefetch_weights(Wh + ((size_t)(Fdim + f)) * (Wdim * Wdim), tid);
    else        prefetch_weights(Wo + ((size_t)f) * Wdim * ncols_out, tid);
    __syncthreads();
    gemm_layer<true>(cur, nxt, WS, bh + (l * Fdim + f) * Wdim, Wdim, wave, lane);
    __syncthreads();
    const unsigned short* t = cur; cur = nxt; nxt = (unsigned short*)t;
  }

  stage_weights(Wo + (size_t)f * Wdim * ncols_out, WS, ncols_out, tid);
  __syncthreads();
  gemm_layer<OUT_BF16>(cur, OutBuf, WS, bo + f * ncols_out, ncols_out, wave, lane);
  __syncthreads();
}

// out[b,o] += sum_k ( sum_w hn[b,w] * arch[o,w,k] ) * g[b,k]
// Phase p: stage arch[p*8 .. p*8+7] (each [W x K], same shape as exp out-weights)
// into WS as 8 swizzled slabs; wave w computes T = Hn @ arch[p*8+w] via WMMA
// (C cols = k), then per-row dot with transposed g via 4 lane shuffles + atomic.
__device__ __forceinline__ void softselect(const unsigned short* __restrict__ Hn,
                                           const float* __restrict__ GT,  // [K][GSTRIDE]
                                           const float* __restrict__ arch,
                                           unsigned* __restrict__ WS,
                                           float* __restrict__ out,
                                           int b0, int wave, int lane, int tid) {
  const int l16 = lane & 15;
  const int kh  = (lane >> 4) * 8;
  const int hi8 = (lane >> 4) * 8;

  for (int phase = 0; phase < 4; ++phase) {
    __syncthreads();   // WS reuse vs previous phase readers
    for (int s = 0; s < 8; ++s)
      stage_weights(arch + (size_t)(phase * 8 + s) * (Wdim * Kdim),
                    WS + s * 2048, Kdim, tid);
    __syncthreads();

    const int o = phase * 8 + wave;
    const unsigned* wlane = WS + wave * 2048 + (lane << 3);  // base < 64KB: immediates
    for (int rt = 0; rt < 8; ++rt) {
      const unsigned short* arow = Hn + (rt * 16 + l16) * Wdim;
      f32x8 C = {0.f, 0.f, 0.f, 0.f, 0.f, 0.f, 0.f, 0.f};
#pragma unroll
      for (int kt = 0; kt < 8; ++kt) {
        ABfrag Af, Bf;
        Af.q[0] = *(const uint4*)(arow + kt * 32 + kh);
        Af.q[1] = *(const uint4*)(arow + kt * 32 + 16 + kh);
        const uint4* wp = (const uint4*)(wlane + (kt << 8));
        Bf.q[0] = wp[0];
        Bf.q[1] = wp[1];
        C = __builtin_amdgcn_wmma_f32_16x16x32_bf16(false, Af.v, false, Bf.v,
                                                    (short)0, C, false, false);
      }
      // This lane's 8 g values: rows rt*16+hi8+0..7 at k = l16 (two b128 loads).
      const float* gp = GT + l16 * GSTRIDE + rt * 16 + hi8;
      const float4 g0 = *(const float4*)(gp);
      const float4 g1 = *(const float4*)(gp + 4);
      const float gv[8] = {g0.x, g0.y, g0.z, g0.w, g1.x, g1.y, g1.z, g1.w};
      // C element (v, lane): row = rt*16 + hi8 + v, col k = lane%16.
#pragma unroll
      for (int v = 0; v < 8; ++v) {
        float p = C[v] * gv[v];
        p += __shfl_xor(p, 1, 32);
        p += __shfl_xor(p, 2, 32);
        p += __shfl_xor(p, 4, 32);
        p += __shfl_xor(p, 8, 32);   // sum over the 16 k-lanes of this half
        if (l16 == v)
          atomicAdd(&out[(size_t)(b0 + rt * 16 + hi8 + v) * Odim + o], p);
      }
    }
  }
}

// LDS layout: WS 128K | HA 64K | HB 64K | GT 16*132*4  = 270592 B (<= 320K WGP LDS)
#define SMEM_BYTES (131072 + 65536 + 65536 + (Kdim * GSTRIDE * 4))

__global__ __launch_bounds__(NTH)
void ngamoe_fused_kernel(const float* __restrict__ x,
                         const float* __restrict__ Wi_n, const float* __restrict__ bi_n,
                         const float* __restrict__ Wh_n, const float* __restrict__ bh_n,
                         const float* __restrict__ Wo_n, const float* __restrict__ bo_n,
                         const float* __restrict__ Wi_e, const float* __restrict__ bi_e,
                         const float* __restrict__ Wh_e, const float* __restrict__ bh_e,
                         const float* __restrict__ Wo_e, const float* __restrict__ bo_e,
                         const float* __restrict__ arch,
                         float* __restrict__ out) {
  extern __shared__ unsigned char smem[];
  unsigned*       WS = (unsigned*)(smem);                 // B fragments (hot) at low offsets
  unsigned short* HA = (unsigned short*)(smem + 131072);
  unsigned short* HB = (unsigned short*)(smem + 196608);
  float*          GT = (float*)(smem + 262144);

  const int tid  = threadIdx.x;
  const int wave = tid >> 5;
  const int lane = tid & 31;
  const int f    = blockIdx.y;
  const int b0   = blockIdx.x * BT;
  const float* xcol = x + (size_t)b0 * Fdim + f;

  // 1) expert path -> g in GT (f32, transposed [K][GSTRIDE])
  run_path<false>(xcol, Wi_e, bi_e, Wh_e, bh_e, Wo_e, bo_e, Kdim,
                  HA, HB, WS, GT, f, tid, wave, lane);
  // 2) nam path -> hn in HB (bf16 [BT x W]); chain: HA->HB->HA->HB
  run_path<true>(xcol, Wi_n, bi_n, Wh_n, bh_n, Wo_n, bo_n, Wdim,
                 HA, HB, WS, HB, f, tid, wave, lane);
  // 3) SoftSelect partial sums into out (atomic over features)
  softselect(HB, GT, arch, WS, out, b0, wave, lane, tid);
}

extern "C" void kernel_launch(void* const* d_in, const int* in_sizes, int n_in,
                              void* d_out, int out_size, void* d_ws, size_t ws_size,
                              hipStream_t stream) {
  (void)in_sizes; (void)n_in; (void)d_ws; (void)ws_size;
  const float* x    = (const float*)d_in[0];
  const float* Wi_n = (const float*)d_in[1];
  const float* bi_n = (const float*)d_in[2];
  const float* Wh_n = (const float*)d_in[3];
  const float* bh_n = (const float*)d_in[4];
  const float* Wo_n = (const float*)d_in[5];
  const float* bo_n = (const float*)d_in[6];
  const float* Wi_e = (const float*)d_in[7];
  const float* bi_e = (const float*)d_in[8];
  const float* Wh_e = (const float*)d_in[9];
  const float* bh_e = (const float*)d_in[10];
  const float* Wo_e = (const float*)d_in[11];
  const float* bo_e = (const float*)d_in[12];
  const float* arch = (const float*)d_in[13];
  float* out = (float*)d_out;

  // Output is accumulated with atomics; zero it first (capturable memset node).
  hipMemsetAsync(d_out, 0, (size_t)out_size * sizeof(float), stream);

  dim3 grid(Bsz / BT, Fdim);
  ngamoe_fused_kernel<<<grid, NTH, SMEM_BYTES, stream>>>(
      x, Wi_n, bi_n, Wh_n, bh_n, Wo_n, bo_n,
      Wi_e, bi_e, Wh_e, bh_e, Wo_e, bo_e, arch, out);
}
